// MSAColumnGlobalAttention_18279380812220
// MI455X (gfx1250) — matrix operand order repaired and checked by reading, not compiled, not run on hardware
//
#include <hip/hip_runtime.h>
#include <hip/hip_bf16.h>

typedef _Float16 f16;
typedef __attribute__((ext_vector_type(16))) _Float16 v16h;
typedef __attribute__((ext_vector_type(8)))  float    v8f;

typedef unsigned int u32x4 __attribute__((ext_vector_type(4)));
typedef int          i32x4 __attribute__((ext_vector_type(4)));
typedef int          i32x8 __attribute__((ext_vector_type(8)));

#define S_DIM 2048
#define R_DIM 512
#define C_DIM 64
#define NTHREADS 512
#define NWAVES 16
#define TILES_PER_WAVE 8   // 128 tiles of 16 rows / 16 waves
#define CHUNK_ROWS 64
#define NCHUNKS 32         // 2048 / 64

// ---- LDS layout (bytes) ----
#define OFF_XLN     0            // f16 [2048*64]  = 262144
#define OFF_LOGITS  262144       // f16 [8*2048]   =  32768  (Phase A: 2x16KB TDM staging)
#define OFF_MASKF   294912       // f16 [2048]     =   4096
#define OFF_BEFF    299008       // f16 [64*16]    =   2048
#define OFF_QPOOL   301056       // f32 [64]
#define OFF_Q       301312       // f32 [64]
#define OFF_ONUM    301568       // f32 [64]
#define OFF_OFINAL  301824       // f32 [64]
#define OFF_RED     302080       // f32 [512]      =   2048
#define OFF_SUMH    304128       // f32 [8]
#define OFF_MAXH    304160       // f32 [8]
#define OFF_MSKSUM  304192       // f32 [1]
#define SMEM_BYTES  304256       // < 320 KB WGP LDS

#if defined(__has_builtin)
#if __has_builtin(__builtin_amdgcn_tensor_load_to_lds) && __has_builtin(__builtin_amdgcn_s_wait_tensorcnt)
#define HAVE_TDM 1
#endif
#endif
#ifndef HAVE_TDM
#define HAVE_TDM 0
#endif

#define WMMA_F16(acc, a, b) \
  acc = __builtin_amdgcn_wmma_f32_16x16x32_f16(false, (a), false, (b), (short)0, (acc), false, false)

// A operand: 16x32 f16 tile, rows s0..s0+15, K-block [k0,k0+32).
// lanes 0-15: M=lane,   elems 0..15 -> K=k0+e
// lanes 16-31: M=lane-16, elems 0..15 -> K=k0+16+e
__device__ __forceinline__ v16h load_a_tile(const f16* __restrict__ x, int s0, int k0, int lane) {
  const int m     = lane & 15;
  const int half8 = (lane >> 4) * 16;
  const f16* rowp = x + (s0 + m) * C_DIM;
  v16h a;
#pragma unroll
  for (int e = 0; e < 8; ++e) a[e]     = rowp[k0 + half8 + e];
#pragma unroll
  for (int e = 0; e < 8; ++e) a[8 + e] = rowp[k0 + 8 + half8 + e];
  return a;
}

__global__ void __launch_bounds__(NTHREADS)
msa_col_global_attn(const float* __restrict__ m_in, const float* __restrict__ mask,
                    const float* __restrict__ ln_w, const float* __restrict__ ln_b,
                    const float* __restrict__ wq,   const float* __restrict__ wk,
                    const float* __restrict__ wv,   const float* __restrict__ wg,
                    const float* __restrict__ bg,   const float* __restrict__ wo,
                    const float* __restrict__ bo,   float* __restrict__ out)
{
  extern __shared__ char smem[];
  f16*   xln     = (f16*)(smem + OFF_XLN);
  f16*   logitsL = (f16*)(smem + OFF_LOGITS);
  f16*   maskf   = (f16*)(smem + OFF_MASKF);
  f16*   beffL   = (f16*)(smem + OFF_BEFF);
  float* qpool   = (float*)(smem + OFF_QPOOL);
  float* qh      = (float*)(smem + OFF_Q);
  float* onum    = (float*)(smem + OFF_ONUM);
  float* ofinal  = (float*)(smem + OFF_OFINAL);
  float* red     = (float*)(smem + OFF_RED);
  float* sumh    = (float*)(smem + OFF_SUMH);
  float* maxh    = (float*)(smem + OFF_MAXH);
  float* msksum  = (float*)(smem + OFF_MSKSUM);

  const int tid  = threadIdx.x;
  const int lane = tid & 31;
  const int wave = tid >> 5;
  const int r    = blockIdx.x;

  if (tid < 64) { qpool[tid] = 0.0f; onum[tid] = 0.0f; }
  if (tid == 0) msksum[0] = 0.0f;
  __syncthreads();

  // ===== Phase A: m[:,s,r,:] -> layernorm -> LDS f16; masked q_pool =========
  {
    const int l16 = tid & 15;      // 4 columns per lane
    const int rg  = tid >> 4;      // 32 rows in flight
    const float4 lw = *(const float4*)(ln_w + l16 * 4);
    const float4 lb = *(const float4*)(ln_b + l16 * 4);
    float qp0 = 0.f, qp1 = 0.f, qp2 = 0.f, qp3 = 0.f, msl = 0.f;

#if HAVE_TDM
    // --- TDM double-buffered staging: 2 x (64 rows x 64 f32) in logits region
    float* stage = (float*)(smem + OFF_LOGITS);
    const unsigned ldsbase = (unsigned)(size_t)(void*)stage;

    auto tdm_issue = [&](int chunk) {
      const unsigned buf = (unsigned)(chunk & 1);
      const unsigned long long ga =
          (unsigned long long)(size_t)(m_in + ((size_t)(chunk * CHUNK_ROWS) * R_DIM + r) * C_DIM);
      u32x4 g0;
      g0[0] = 1u;                                              // count=1, load, user D#
      g0[1] = ldsbase + buf * (CHUNK_ROWS * C_DIM * 4u);       // lds_addr (bytes)
      g0[2] = (unsigned)(ga & 0xffffffffull);                  // global_addr[31:0]
      g0[3] = (unsigned)((ga >> 32) & 0x1ffffffull) | (2u << 30); // addr[56:32] | type=2
      i32x8 g1;
      g1[0] = (2 << 16);           // workgroup_mask=0, data_size=2 (4 bytes)
      g1[1] = (C_DIM << 16);       // tensor_dim0[15:0] = 64
      g1[2] = (S_DIM << 16);       // tensor_dim0 hi = 0; tensor_dim1[15:0] = 2048
      g1[3] = (C_DIM << 16);       // tensor_dim1 hi = 0; tile_dim0 = 64
      g1[4] = CHUNK_ROWS;          // tile_dim1 = 64; tile_dim2 = 0
      g1[5] = R_DIM * C_DIM;       // tensor_dim0_stride = 32768 elements
      g1[6] = 0;                   // stride hi; tensor_dim1_stride lo (unused, 2D)
      g1[7] = 0;
      i32x4 g2 = {0, 0, 0, 0};
      i32x4 g3 = {0, 0, 0, 0};
#if __clang_major__ >= 23
      i32x8 g4 = {0, 0, 0, 0, 0, 0, 0, 0};
      __builtin_amdgcn_tensor_load_to_lds(g0, g1, g2, g3, g4, 0);
#else
      __builtin_amdgcn_tensor_load_to_lds(g0, g1, g2, g3, 0);
#endif
    };

    for (int chunk = 0; chunk < NCHUNKS; ++chunk) {
      if (wave == 0) {
        if (chunk == 0) { tdm_issue(0); tdm_issue(1); }
        if (chunk < NCHUNKS - 1) __builtin_amdgcn_s_wait_tensorcnt(1);
        else                     __builtin_amdgcn_s_wait_tensorcnt(0);
      }
      __syncthreads();   // staged chunk visible to all waves
      const float* bufp = stage + (size_t)(chunk & 1) * (CHUNK_ROWS * C_DIM);
#pragma unroll
      for (int hh = 0; hh < 2; ++hh) {
        const int sl = rg + 32 * hh;             // row within chunk
        const int s  = chunk * CHUNK_ROWS + sl;  // global row
        const float4 v = *(const float4*)(bufp + sl * C_DIM + l16 * 4);
        float ps = v.x + v.y + v.z + v.w;
        float pq = v.x * v.x + v.y * v.y + v.z * v.z + v.w * v.w;
#pragma unroll
        for (int d = 1; d < 16; d <<= 1) {
          ps += __shfl_xor(ps, d, 32);
          pq += __shfl_xor(pq, d, 32);
        }
        const float mu  = ps * (1.0f / 64.0f);
        const float var = pq * (1.0f / 64.0f) - mu * mu;
        const float rs  = rsqrtf(var + 1e-5f);
        const float x0 = (v.x - mu) * rs * lw.x + lb.x;
        const float x1 = (v.y - mu) * rs * lw.y + lb.y;
        const float x2 = (v.z - mu) * rs * lw.z + lb.z;
        const float x3 = (v.w - mu) * rs * lw.w + lb.w;
        const float mk = mask[(size_t)s * R_DIM + r];
        qp0 += x0 * mk; qp1 += x1 * mk; qp2 += x2 * mk; qp3 += x3 * mk;
        f16* xp = xln + s * C_DIM + l16 * 4;
        xp[0] = (f16)x0; xp[1] = (f16)x1; xp[2] = (f16)x2; xp[3] = (f16)x3;
        if (l16 == 0) { maskf[s] = (f16)mk; msl += mk; }
      }
      __syncthreads();   // chunk consumed; its buffer may be overwritten
      if (wave == 0 && chunk + 2 < NCHUNKS) tdm_issue(chunk + 2);
    }
#else
    // --- Fallback: direct coalesced global loads ---
    for (int it = 0; it < 64; ++it) {
      const int s = rg + 32 * it;
      const float4 v = *(const float4*)(m_in + ((size_t)s * R_DIM + r) * C_DIM + l16 * 4);
      float ps = v.x + v.y + v.z + v.w;
      float pq = v.x * v.x + v.y * v.y + v.z * v.z + v.w * v.w;
#pragma unroll
      for (int d = 1; d < 16; d <<= 1) {
        ps += __shfl_xor(ps, d, 32);
        pq += __shfl_xor(pq, d, 32);
      }
      const float mu  = ps * (1.0f / 64.0f);
      const float var = pq * (1.0f / 64.0f) - mu * mu;
      const float rs  = rsqrtf(var + 1e-5f);
      const float x0 = (v.x - mu) * rs * lw.x + lb.x;
      const float x1 = (v.y - mu) * rs * lw.y + lb.y;
      const float x2 = (v.z - mu) * rs * lw.z + lb.z;
      const float x3 = (v.w - mu) * rs * lw.w + lb.w;
      const float mk = mask[(size_t)s * R_DIM + r];
      qp0 += x0 * mk; qp1 += x1 * mk; qp2 += x2 * mk; qp3 += x3 * mk;
      f16* xp = xln + s * C_DIM + l16 * 4;
      xp[0] = (f16)x0; xp[1] = (f16)x1; xp[2] = (f16)x2; xp[3] = (f16)x3;
      if (l16 == 0) { maskf[s] = (f16)mk; msl += mk; }
    }
#endif
    atomicAdd(&qpool[l16 * 4 + 0], qp0);
    atomicAdd(&qpool[l16 * 4 + 1], qp1);
    atomicAdd(&qpool[l16 * 4 + 2], qp2);
    atomicAdd(&qpool[l16 * 4 + 3], qp3);
    if (l16 == 0) atomicAdd(&msksum[0], msl);
  }
  __syncthreads();

  // ===== Phase B: q = (q_pool/n) @ wq * ch^-0.5; B_eff = [wk.q | wv] ========
  if (tid < 64) {
    const float inv = 1.0f / (msksum[0] + 1e-10f);
    float acc = 0.0f;
    for (int c = 0; c < 64; ++c) acc += qpool[c] * wq[c * 64 + tid];
    qh[tid] = acc * inv * 0.35355339059327373f;   // 1/sqrt(C_HID=8)
  }
  __syncthreads();
  {
    const int c = tid >> 3, h = tid & 7;
    float we = 0.0f;
#pragma unroll
    for (int ch = 0; ch < 8; ++ch) we += wk[c * 8 + ch] * qh[h * 8 + ch];
    beffL[c * 16 + h]     = (f16)we;              // cols 0..7  -> logits
    beffL[c * 16 + 8 + h] = (f16)wv[c * 8 + h];   // cols 8..15 -> v
  }
  __syncthreads();

  const int col   = lane & 15;
  const int half8 = (lane >> 4) * 8;
  const int khalf = (lane >> 4) * 16;

  // B operand regs: lane n holds column N=n&15, K = khalf+e (+32 per step)
  v16h beff0, beff1;
#pragma unroll
  for (int e = 0; e < 16; ++e) {
    beff0[e] = beffL[(khalf + e) * 16 + col];
    beff1[e] = beffL[(32 + khalf + e) * 16 + col];
  }

  // ===== Phase C: logits via WMMA, store f16 (+mask bias) ===================
  for (int tt = 0; tt < TILES_PER_WAVE; ++tt) {
    const int s0 = (wave + NWAVES * tt) * 16;
    const v16h a0 = load_a_tile(xln, s0, 0, lane);
    const v16h a1 = load_a_tile(xln, s0, 32, lane);
    v8f acc = {};
    WMMA_F16(acc, a0, beff0);
    WMMA_F16(acc, a1, beff1);
    if (col < 8) {
#pragma unroll
      for (int i = 0; i < 8; ++i) {
        const int s = s0 + half8 + i;
        const float b = 1e9f * ((float)maskf[s] - 1.0f);
        logitsL[col * S_DIM + s] = (f16)(acc[i] + b);   // -1e9 -> f16 -inf, exp->0
      }
    }
  }
  __syncthreads();

  // ===== Phase D: softmax stats; exp in place ===============================
  {
    const int h = tid >> 6, tt6 = tid & 63;
    float mx = -1e30f;
    for (int s = tt6; s < S_DIM; s += 64) mx = fmaxf(mx, (float)logitsL[h * S_DIM + s]);
    red[h * 64 + tt6] = mx;
    __syncthreads();
    if (tid < 8) {
      float mm = -1e30f;
      for (int j = 0; j < 64; ++j) mm = fmaxf(mm, red[tid * 64 + j]);
      maxh[tid] = mm;
    }
    __syncthreads();
    const float mh = maxh[h];
    float sl = 0.0f;
    for (int s = tt6; s < S_DIM; s += 64) {
      const float a = __expf((float)logitsL[h * S_DIM + s] - mh);
      logitsL[h * S_DIM + s] = (f16)a;
      sl += a;
    }
    red[h * 64 + tt6] = sl;
    __syncthreads();
    if (tid < 8) {
      float ss = 0.0f;
      for (int j = 0; j < 64; ++j) ss += red[tid * 64 + j];
      sumh[tid] = ss;
    }
  }
  __syncthreads();

  // ===== Phase E: o = a @ v (v regenerated via WMMA) ========================
  {
    float op[8] = {0.f, 0.f, 0.f, 0.f, 0.f, 0.f, 0.f, 0.f};
    for (int tt = 0; tt < TILES_PER_WAVE; ++tt) {
      const int s0 = (wave + NWAVES * tt) * 16;
      const v16h a0 = load_a_tile(xln, s0, 0, lane);
      const v16h a1 = load_a_tile(xln, s0, 32, lane);
      v8f acc = {};
      WMMA_F16(acc, a0, beff0);
      WMMA_F16(acc, a1, beff1);
      if (col >= 8) {                 // cols 8..15 hold v[s][ch]
#pragma unroll
        for (int h = 0; h < 8; ++h) {
          float oa = 0.0f;
#pragma unroll
          for (int i = 0; i < 8; ++i)
            oa += (float)logitsL[h * S_DIM + s0 + half8 + i] * acc[i];
          op[h] += oa;
        }
      }
    }
    if (col >= 8) {
#pragma unroll
      for (int h = 0; h < 8; ++h) atomicAdd(&onum[h * 8 + (col - 8)], op[h]);
    }
  }
  __syncthreads();
  if (tid < 64) ofinal[tid] = onum[tid] / sumh[tid >> 3];
  __syncthreads();

  // ===== Phase F1: g = sigmoid(x@wg + bg); gate by o; in-place f16 ==========
  {
    v16h wgB[8];
#pragma unroll
    for (int j = 0; j < 4; ++j)
#pragma unroll
      for (int t = 0; t < 2; ++t) {
        v16h bb;
#pragma unroll
        for (int e = 0; e < 16; ++e) {
          const int k = 32 * t + khalf + e;
          bb[e] = (f16)wg[k * 64 + 16 * j + col];
        }
        wgB[j * 2 + t] = bb;
      }
    for (int tt = 0; tt < TILES_PER_WAVE; ++tt) {
      const int s0 = (wave + NWAVES * tt) * 16;
      const v16h a0 = load_a_tile(xln, s0, 0, lane);
      const v16h a1 = load_a_tile(xln, s0, 32, lane);
#pragma unroll
      for (int j = 0; j < 4; ++j) {
        v8f acc = {};
        WMMA_F16(acc, a0, wgB[2 * j]);
        WMMA_F16(acc, a1, wgB[2 * j + 1]);
        const int cc  = 16 * j + col;
        const float bgv = bg[cc];
        const float ofv = ofinal[cc];
#pragma unroll
        for (int i = 0; i < 8; ++i) {
          const float gg = 1.0f / (1.0f + __expf(-(acc[i] + bgv)));
          xln[(s0 + half8 + i) * C_DIM + cc] = (f16)(gg * ofv);
        }
      }
    }
  }
  __syncthreads();

  // ===== Phase F2: out = gated @ wo + bo -> global [S,R,C] ==================
  {
    v16h woB[8];
#pragma unroll
    for (int j = 0; j < 4; ++j)
#pragma unroll
      for (int t = 0; t < 2; ++t) {
        v16h bb;
#pragma unroll
        for (int e = 0; e < 16; ++e) {
          const int k = 32 * t + khalf + e;
          bb[e] = (f16)wo[k * 64 + 16 * j + col];
        }
        woB[j * 2 + t] = bb;
      }
    for (int tt = 0; tt < TILES_PER_WAVE; ++tt) {
      const int s0 = (wave + NWAVES * tt) * 16;
      const v16h a0 = load_a_tile(xln, s0, 0, lane);
      const v16h a1 = load_a_tile(xln, s0, 32, lane);
#pragma unroll
      for (int j = 0; j < 4; ++j) {
        v8f acc = {};
        WMMA_F16(acc, a0, woB[2 * j]);
        WMMA_F16(acc, a1, woB[2 * j + 1]);
        const int cc  = 16 * j + col;
        const float bov = bo[cc];
#pragma unroll
        for (int i = 0; i < 8; ++i) {
          const int s = s0 + half8 + i;
          out[((size_t)s * R_DIM + r) * C_DIM + cc] = acc[i] + bov;
        }
      }
    }
  }
}

extern "C" void kernel_launch(void* const* d_in, const int* in_sizes, int n_in,
                              void* d_out, int out_size, void* d_ws, size_t ws_size,
                              hipStream_t stream) {
  (void)in_sizes; (void)n_in; (void)out_size; (void)d_ws; (void)ws_size;
  const float* m_in = (const float*)d_in[0];
  const float* mask = (const float*)d_in[1];
  const float* ln_w = (const float*)d_in[2];
  const float* ln_b = (const float*)d_in[3];
  const float* wq   = (const float*)d_in[4];
  const float* wk   = (const float*)d_in[5];
  const float* wv   = (const float*)d_in[6];
  const float* wg   = (const float*)d_in[7];
  const float* bg   = (const float*)d_in[8];
  const float* wo   = (const float*)d_in[9];
  const float* bo   = (const float*)d_in[10];
  float* out = (float*)d_out;

  msa_col_global_attn<<<dim3(R_DIM), dim3(NTHREADS), SMEM_BYTES, stream>>>(
      m_in, mask, ln_w, ln_b, wq, wk, wv, wg, bg, wo, bo, out);
}
